// AdaAttN_16475494548240
// MI455X (gfx1250) — matrix operand order repaired and checked by reading, not compiled, not run on hardware
//
#include <hip/hip_runtime.h>
#include <cstdint>
#include <cstddef>

// ---------------------------------------------------------------------------
// AdaAttN on MI455X (gfx1250) — bf16 WMMA pipeline, f32 accumulation.
// ~232 GFLOP vs ~160 MB input -> matrix-math bound: all GEMMs run on
// v_wmma_f32_16x16x32_bf16. GEMM staging uses CDNA5 async loads to LDS
// (global_load_async_to_lds_b128, ASYNCcnt) with double-buffered tiles so
// WMMA of tile i overlaps the fill of tile i+1. All epilogues are contiguous
// b128 stores (operand roles chosen so per-lane M-runs hit contiguous memory).
//
// Workspace (assumes ws_size >= ~250 MB).
// ---------------------------------------------------------------------------

typedef __bf16 bf16_t;
typedef __attribute__((ext_vector_type(16))) __bf16 v16bf;
typedef __attribute__((ext_vector_type(8)))  __bf16 v8bf;
typedef __attribute__((ext_vector_type(8)))  float  v8f;
typedef __attribute__((ext_vector_type(4)))  float  v4f;

namespace {
constexpr int   BATCH = 4;
constexpr int   CH    = 512;    // V == QK
constexpr int   LL    = 4096;   // H*W
constexpr float EPSV  = 1e-5f;
constexpr int   PITCH = 72;     // LDS row pitch in bf16 (144 B, 16B-aligned)
}

// --- CDNA5 helpers ---------------------------------------------------------
__device__ __forceinline__ v8f wmma_bf16(v16bf a, v16bf b, v8f c) {
    return __builtin_amdgcn_wmma_f32_16x16x32_bf16(false, a, false, b,
                                                   (short)0, c, false, false);
}
// A operand (16x32): lane<16 -> row m, K{0..7,16..23}; lane>=16 -> K{8..15,24..31}
__device__ __forceinline__ v16bf load_a(const bf16_t* row, int lane) {
    const int sel = (lane >> 4) * 8;
    v8bf a0 = *(const v8bf*)(row + sel);
    v8bf a1 = *(const v8bf*)(row + sel + 16);
    return __builtin_shufflevector(a0, a1, 0,1,2,3,4,5,6,7,8,9,10,11,12,13,14,15);
}
// B operand (32x16): lane<16 -> col n, K{0..7,16..23}; lane>=16 -> K{8..15,24..31}
__device__ __forceinline__ v16bf load_b(const bf16_t* row, int lane) {
    const int sel = (lane >> 4) * 8;
    v8bf b0 = *(const v8bf*)(row + sel);
    v8bf b1 = *(const v8bf*)(row + sel + 16);
    return __builtin_shufflevector(b0, b1, 0,1,2,3,4,5,6,7,8,9,10,11,12,13,14,15);
}
// Async global->LDS 16B copy (ASYNCcnt). Low 32 bits of a generic shared
// pointer are the LDS byte address (LDS aperture: addr[31:0] = offset).
__device__ __forceinline__ void async_copy_b128(void* lds_ptr, const void* gptr) {
    uint32_t lds = (uint32_t)(uintptr_t)lds_ptr;
    asm volatile("global_load_async_to_lds_b128 %0, %1, off"
                 :: "v"(lds), "v"(gptr) : "memory");
}
__device__ __forceinline__ void wait_async() {
    asm volatile("s_wait_asynccnt 0x0" ::: "memory");
}

// ---------------------------------------------------------------------------
// Per-(batch,channel) mean / rstd (unbiased var + eps). grid = BATCH*CH
// ---------------------------------------------------------------------------
__global__ __launch_bounds__(256, 1)
void stats_kernel(const float* __restrict__ x, float* __restrict__ mr) {
    const int ch = blockIdx.x;
    const float* p = x + (size_t)ch * LL;
    float s = 0.f, ss = 0.f;
    for (int i = threadIdx.x; i < LL; i += 256) {
        float v = p[i];
        s += v; ss += v * v;
    }
    __shared__ float rs[256], rss[256];
    rs[threadIdx.x] = s; rss[threadIdx.x] = ss;
    __syncthreads();
    for (int off = 128; off > 0; off >>= 1) {
        if (threadIdx.x < off) {
            rs[threadIdx.x]  += rs[threadIdx.x + off];
            rss[threadIdx.x] += rss[threadIdx.x + off];
        }
        __syncthreads();
    }
    if (threadIdx.x == 0) {
        float m   = rs[0] / (float)LL;
        float var = (rss[0] - rs[0] * m) / (float)(LL - 1);
        mr[2 * ch]     = m;
        mr[2 * ch + 1] = rsqrtf(var + EPSV);
    }
}

// ---------------------------------------------------------------------------
// f32 -> bf16 weight conversion
// ---------------------------------------------------------------------------
__global__ __launch_bounds__(256, 1)
void cvt_bf16_kernel(const float* __restrict__ w, bf16_t* __restrict__ wb, int n) {
    int i = blockIdx.x * 256 + threadIdx.x;
    if (i < n) wb[i] = (bf16_t)w[i];
}

// ---------------------------------------------------------------------------
// Transpose (+ optional mvn) to bf16: X[B][CH][LL] f32 -> Out[B][LL][CH] bf16
// 64x64 tiles via LDS. grid = (LL/64, CH/64, BATCH)
// ---------------------------------------------------------------------------
template<bool NORM>
__global__ __launch_bounds__(256, 1)
void transpose_kernel(const float* __restrict__ X, const float* __restrict__ stats,
                      bf16_t* __restrict__ Out) {
    const int b     = blockIdx.z;
    const int lBase = blockIdx.x * 64;
    const int cBase = blockIdx.y * 64;
    const float* Xb = X + (size_t)b * CH * LL;
    bf16_t* Ob      = Out + (size_t)b * LL * CH;

    __shared__ bf16_t T[64][PITCH];
    __shared__ float  stl[128];

    const int tid = threadIdx.x;
    if (NORM && tid < 128)
        stl[tid] = stats[(size_t)b * CH * 2 + 2 * cBase + tid];
    __syncthreads();

    #pragma unroll
    for (int i = 0; i < 4; ++i) {
        int idx = tid + 256 * i;               // 1024 v4f chunks (64c x 16)
        int cc = idx >> 4, lq = idx & 15;
        v4f x = *(const v4f*)(Xb + (size_t)(cBase + cc) * LL + lBase + lq * 4);
        float mean = 0.f, rstd = 1.f;
        if (NORM) { mean = stl[2 * cc]; rstd = stl[2 * cc + 1]; }
        #pragma unroll
        for (int e = 0; e < 4; ++e)
            T[lq * 4 + e][cc] = (bf16_t)((x[e] - mean) * rstd);
    }
    __syncthreads();
    #pragma unroll
    for (int i = 0; i < 2; ++i) {
        int idx = tid + 256 * i;               // 512 v8bf chunks
        int ch = idx & 7, ll = idx >> 3;
        v8bf t = *(const v8bf*)(&T[ll][ch * 8]);
        *(v8bf*)(Ob + (size_t)(lBase + ll) * CH + cBase + ch * 8) = t;
    }
}

// ---------------------------------------------------------------------------
// Q / K projection: Out[l][o] = sum_c W[o][c] * Xt[l][c] + bias[o]
// A = W (M=o), B = Xt (N=l). Double-buffered async staging.
// grid = (LL/128, CH/64, BATCH)
// ---------------------------------------------------------------------------
__global__ __launch_bounds__(256, 1)
void proj_lc_kernel(const bf16_t* __restrict__ Wb, const bf16_t* __restrict__ Xt,
                    const float* __restrict__ bias, bf16_t* __restrict__ Out) {
    const int b     = blockIdx.z;
    const int lBase = blockIdx.x * 128;
    const int oBase = blockIdx.y * 64;
    const bf16_t* Xb = Xt + (size_t)b * LL * CH;
    bf16_t* Ob       = Out + (size_t)b * LL * CH;

    __shared__ bf16_t Ws[2][64][PITCH];
    __shared__ bf16_t Xs[2][128][PITCH];

    const int tid  = threadIdx.x;
    const int wid  = tid >> 5;
    const int lane = tid & 31;
    const int wo   = wid & 1;     // 32 o (M)
    const int wl   = wid >> 1;    // 32 l (N)

    auto stage = [&](int buf, int c0) {
        #pragma unroll
        for (int i = 0; i < 2; ++i) {
            int idx = tid + 256 * i;
            int chk = idx & 7, oo = idx >> 3;
            async_copy_b128(&Ws[buf][oo][chk * 8],
                            Wb + (size_t)(oBase + oo) * CH + c0 + chk * 8);
        }
        #pragma unroll
        for (int i = 0; i < 4; ++i) {
            int idx = tid + 256 * i;
            int chk = idx & 7, ll = idx >> 3;
            async_copy_b128(&Xs[buf][ll][chk * 8],
                            Xb + (size_t)(lBase + ll) * CH + c0 + chk * 8);
        }
    };

    v8f acc[2][2] = {};
    stage(0, 0);
    wait_async();
    __syncthreads();

    for (int c0 = 0, buf = 0; c0 < CH; c0 += 64, buf ^= 1) {
        if (c0 + 64 < CH) stage(buf ^ 1, c0 + 64);
        #pragma unroll
        for (int ks = 0; ks < 2; ++ks) {
            const int base = ks * 32;
            v16bf a[2];
            #pragma unroll
            for (int mt = 0; mt < 2; ++mt)
                a[mt] = load_a(&Ws[buf][32 * wo + 16 * mt + (lane & 15)][base], lane);
            #pragma unroll
            for (int nt = 0; nt < 2; ++nt) {
                v16bf bm = load_b(&Xs[buf][32 * wl + 16 * nt + (lane & 15)][base], lane);
                #pragma unroll
                for (int mt = 0; mt < 2; ++mt)
                    acc[mt][nt] = wmma_bf16(a[mt], bm, acc[mt][nt]);
            }
        }
        wait_async();
        __syncthreads();
    }

    const int n  = lane & 15;
    const int mo = (lane >> 4) * 8;
    #pragma unroll
    for (int nt = 0; nt < 2; ++nt) {
        int l = lBase + 32 * wl + 16 * nt + n;
        #pragma unroll
        for (int mt = 0; mt < 2; ++mt) {
            int o0 = oBase + 32 * wo + 16 * mt + mo;
            v8bf pk;
            #pragma unroll
            for (int r = 0; r < 8; ++r)
                pk[r] = (bf16_t)(acc[mt][nt][r] + bias[o0 + r]);
            *(v8bf*)(Ob + (size_t)l * CH + o0) = pk;
        }
    }
}

// ---------------------------------------------------------------------------
// V projection to [v][k] layout, producing V and V^2 (both bf16):
//   Vh[v][k] = sum_c Xt[k][c] * W[v][c] + bias[v];  Vq = Vh^2
// A = Xt (M=k), B = W (N=v). Double-buffered async staging.
// grid = (LL/128, CH/64, BATCH)
// ---------------------------------------------------------------------------
__global__ __launch_bounds__(256, 1)
void proj_cl_kernel(const bf16_t* __restrict__ Xt, const bf16_t* __restrict__ Wb,
                    const float* __restrict__ bias,
                    bf16_t* __restrict__ OutV, bf16_t* __restrict__ OutVq) {
    const int b     = blockIdx.z;
    const int kBase = blockIdx.x * 128;
    const int vBase = blockIdx.y * 64;
    const bf16_t* Xb = Xt + (size_t)b * LL * CH;
    bf16_t* OVb  = OutV  + (size_t)b * CH * LL;
    bf16_t* OQb  = OutVq + (size_t)b * CH * LL;

    __shared__ bf16_t Xs[2][128][PITCH];
    __shared__ bf16_t Ws[2][64][PITCH];

    const int tid  = threadIdx.x;
    const int wid  = tid >> 5;
    const int lane = tid & 31;
    const int wk   = wid & 3;     // 32 k (M)
    const int wv   = wid >> 2;    // 32 v (N)

    auto stage = [&](int buf, int c0) {
        #pragma unroll
        for (int i = 0; i < 4; ++i) {
            int idx = tid + 256 * i;
            int chk = idx & 7, kk = idx >> 3;
            async_copy_b128(&Xs[buf][kk][chk * 8],
                            Xb + (size_t)(kBase + kk) * CH + c0 + chk * 8);
        }
        #pragma unroll
        for (int i = 0; i < 2; ++i) {
            int idx = tid + 256 * i;
            int chk = idx & 7, vv = idx >> 3;
            async_copy_b128(&Ws[buf][vv][chk * 8],
                            Wb + (size_t)(vBase + vv) * CH + c0 + chk * 8);
        }
    };

    v8f acc[2][2] = {};
    stage(0, 0);
    wait_async();
    __syncthreads();

    for (int c0 = 0, buf = 0; c0 < CH; c0 += 64, buf ^= 1) {
        if (c0 + 64 < CH) stage(buf ^ 1, c0 + 64);
        #pragma unroll
        for (int ks = 0; ks < 2; ++ks) {
            const int base = ks * 32;
            v16bf a[2];
            #pragma unroll
            for (int mt = 0; mt < 2; ++mt)
                a[mt] = load_a(&Xs[buf][32 * wk + 16 * mt + (lane & 15)][base], lane);
            #pragma unroll
            for (int nt = 0; nt < 2; ++nt) {
                v16bf bm = load_b(&Ws[buf][32 * wv + 16 * nt + (lane & 15)][base], lane);
                #pragma unroll
                for (int mt = 0; mt < 2; ++mt)
                    acc[mt][nt] = wmma_bf16(a[mt], bm, acc[mt][nt]);
            }
        }
        wait_async();
        __syncthreads();
    }

    const int n  = lane & 15;
    const int mo = (lane >> 4) * 8;
    #pragma unroll
    for (int nt = 0; nt < 2; ++nt) {
        int v = vBase + 32 * wv + 16 * nt + n;
        float bv = bias[v];
        #pragma unroll
        for (int mt = 0; mt < 2; ++mt) {
            int k0r = kBase + 32 * wk + 16 * mt + mo;
            v8bf pk, pq;
            #pragma unroll
            for (int r = 0; r < 8; ++r) {
                float x = acc[mt][nt][r] + bv;
                pk[r] = (bf16_t)x;
                float xr = (float)pk[r];
                pq[r] = (bf16_t)(xr * xr);
            }
            *(v8bf*)(OVb + (size_t)v * LL + k0r) = pk;
            *(v8bf*)(OQb + (size_t)v * LL + k0r) = pq;
        }
    }
}

// ---------------------------------------------------------------------------
// Attention scores: S[q][k] = sum_c Q[q][c] * Kt[k][c]
// A = Kt (M=k), B = Q (N=q). Double-buffered async staging; b128 stores.
// grid = (LL/128 (k), LL/128 (q), BATCH)
// ---------------------------------------------------------------------------
__global__ __launch_bounds__(256, 1)
void attn_qk_kernel(const bf16_t* __restrict__ Q, const bf16_t* __restrict__ Kt,
                    bf16_t* __restrict__ S) {
    const int b     = blockIdx.z;
    const int kBase = blockIdx.x * 128;
    const int qBase = blockIdx.y * 128;
    const bf16_t* Qb = Q  + (size_t)b * LL * CH;
    const bf16_t* Kb = Kt + (size_t)b * LL * CH;

    __shared__ bf16_t Ks[2][128][PITCH];
    __shared__ bf16_t Qs[2][128][PITCH];

    const int tid  = threadIdx.x;
    const int wid  = tid >> 5;
    const int lane = tid & 31;
    const int wk   = wid & 3;     // 32 k (M)
    const int wq   = wid >> 2;    // 64 q (N)

    auto stage = [&](int buf, int c0) {
        #pragma unroll
        for (int i = 0; i < 4; ++i) {
            int idx = tid + 256 * i;
            int chk = idx & 7, rr = idx >> 3;
            async_copy_b128(&Ks[buf][rr][chk * 8],
                            Kb + (size_t)(kBase + rr) * CH + c0 + chk * 8);
            async_copy_b128(&Qs[buf][rr][chk * 8],
                            Qb + (size_t)(qBase + rr) * CH + c0 + chk * 8);
        }
    };

    v8f acc[2][4] = {};
    stage(0, 0);
    wait_async();
    __syncthreads();

    for (int c0 = 0, buf = 0; c0 < CH; c0 += 64, buf ^= 1) {
        if (c0 + 64 < CH) stage(buf ^ 1, c0 + 64);
        #pragma unroll
        for (int ks = 0; ks < 2; ++ks) {
            const int base = ks * 32;
            v16bf a[2];
            #pragma unroll
            for (int mt = 0; mt < 2; ++mt)
                a[mt] = load_a(&Ks[buf][32 * wk + 16 * mt + (lane & 15)][base], lane);
            #pragma unroll
            for (int nt = 0; nt < 4; ++nt) {
                v16bf bm = load_b(&Qs[buf][64 * wq + 16 * nt + (lane & 15)][base], lane);
                #pragma unroll
                for (int mt = 0; mt < 2; ++mt)
                    acc[mt][nt] = wmma_bf16(a[mt], bm, acc[mt][nt]);
            }
        }
        wait_async();
        __syncthreads();
    }

    const int n  = lane & 15;
    const int mo = (lane >> 4) * 8;
    #pragma unroll
    for (int nt = 0; nt < 4; ++nt) {
        int q = qBase + 64 * wq + 16 * nt + n;
        #pragma unroll
        for (int mt = 0; mt < 2; ++mt) {
            int kk0 = kBase + 32 * wk + 16 * mt + mo;
            v8bf pk;
            #pragma unroll
            for (int r = 0; r < 8; ++r)
                pk[r] = (bf16_t)acc[mt][nt][r];
            *(v8bf*)(S + ((size_t)b * LL + q) * LL + kk0) = pk;
        }
    }
}

// ---------------------------------------------------------------------------
// Row softmax: rowM/rowL stats AND in-place S <- exp(S - rowM)  (bf16).
// exp is computed exactly once per element (attn_pv then just copies).
// grid = (LL, BATCH)
// ---------------------------------------------------------------------------
__global__ __launch_bounds__(256, 1)
void row_softmax_kernel(bf16_t* __restrict__ S,
                        float* __restrict__ rowM, float* __restrict__ rowL) {
    const int q = blockIdx.x;
    const int b = blockIdx.y;
    bf16_t* row = S + ((size_t)b * LL + q) * LL;
    v8bf vv[2];
    float m = -3.0e38f;
    #pragma unroll
    for (int j = 0; j < 2; ++j) {
        vv[j] = *(const v8bf*)(row + threadIdx.x * 8 + 2048 * j);
        #pragma unroll
        for (int e = 0; e < 8; ++e) m = fmaxf(m, (float)vv[j][e]);
    }
    __shared__ float red[256];
    red[threadIdx.x] = m;
    __syncthreads();
    for (int off = 128; off > 0; off >>= 1) {
        if (threadIdx.x < off)
            red[threadIdx.x] = fmaxf(red[threadIdx.x], red[threadIdx.x + off]);
        __syncthreads();
    }
    m = red[0];
    __syncthreads();
    float s = 0.f;
    v8bf ev[2];
    #pragma unroll
    for (int j = 0; j < 2; ++j)
        #pragma unroll
        for (int e = 0; e < 8; ++e) {
            float p = __expf((float)vv[j][e] - m);
            ev[j][e] = (bf16_t)p;
            s += p;
        }
    red[threadIdx.x] = s;
    __syncthreads();
    for (int off = 128; off > 0; off >>= 1) {
        if (threadIdx.x < off)
            red[threadIdx.x] += red[threadIdx.x + off];
        __syncthreads();
    }
    #pragma unroll
    for (int j = 0; j < 2; ++j)
        *(v8bf*)(row + threadIdx.x * 8 + 2048 * j) = ev[j];
    if (threadIdx.x == 0) {
        rowM[(size_t)b * LL + q] = m;
        rowL[(size_t)b * LL + q] = red[0];
    }
}

// ---------------------------------------------------------------------------
// P*V and P*V^2 + epilogue: out = sqrt(max(E2-M^2,0))*mvn(c_x) + M
// P is pre-exponentiated (in S); V and V^2 precomputed -> staging is three
// pure async copies, double-buffered. D lane fixed v, r walks q -> b128 f32.
// grid = (LL/128, CH/64, BATCH)
// ---------------------------------------------------------------------------
__global__ __launch_bounds__(256, 1)
void attn_pv_kernel(const bf16_t* __restrict__ P,
                    const float* __restrict__ rowL,
                    const bf16_t* __restrict__ Vh, const bf16_t* __restrict__ Vq,
                    const float* __restrict__ c_x, const float* __restrict__ stats_cx,
                    float* __restrict__ out) {
    const int b     = blockIdx.z;
    const int qBase = blockIdx.x * 128;
    const int vBase = blockIdx.y * 64;
    const bf16_t* Pb  = P  + (size_t)b * LL * LL;
    const bf16_t* Vb  = Vh + (size_t)b * CH * LL;
    const bf16_t* Vqb = Vq + (size_t)b * CH * LL;

    __shared__ bf16_t Pt[2][128][PITCH];
    __shared__ bf16_t Vs[2][64][PITCH];
    __shared__ bf16_t Vsq[2][64][PITCH];

    const int tid  = threadIdx.x;
    const int wid  = tid >> 5;   // q-group of 16
    const int lane = tid & 31;

    auto stage = [&](int buf, int k0) {
        #pragma unroll
        for (int i = 0; i < 4; ++i) {
            int idx = tid + 256 * i;
            int chk = idx & 7, qq = idx >> 3;
            async_copy_b128(&Pt[buf][qq][chk * 8],
                            Pb + (size_t)(qBase + qq) * LL + k0 + chk * 8);
        }
        #pragma unroll
        for (int i = 0; i < 2; ++i) {
            int idx = tid + 256 * i;
            int chk = idx & 7, vr = idx >> 3;
            async_copy_b128(&Vs[buf][vr][chk * 8],
                            Vb + (size_t)(vBase + vr) * LL + k0 + chk * 8);
            async_copy_b128(&Vsq[buf][vr][chk * 8],
                            Vqb + (size_t)(vBase + vr) * LL + k0 + chk * 8);
        }
    };

    v8f accM[4]  = {};
    v8f accM2[4] = {};
    stage(0, 0);
    wait_async();
    __syncthreads();

    for (int k0 = 0, buf = 0; k0 < LL; k0 += 64, buf ^= 1) {
        if (k0 + 64 < LL) stage(buf ^ 1, k0 + 64);
        #pragma unroll
        for (int ks = 0; ks < 2; ++ks) {
            const int base = ks * 32;
            v16bf a = load_a(&Pt[buf][16 * wid + (lane & 15)][base], lane);
            #pragma unroll
            for (int nt = 0; nt < 4; ++nt) {
                v16bf bv = load_b(&Vs[buf][16 * nt + (lane & 15)][base], lane);
                accM[nt] = wmma_bf16(a, bv, accM[nt]);
                v16bf bs = load_b(&Vsq[buf][16 * nt + (lane & 15)][base], lane);
                accM2[nt] = wmma_bf16(a, bs, accM2[nt]);
            }
        }
        wait_async();
        __syncthreads();
    }

    const int n     = lane & 15;
    const int mo    = (lane >> 4) * 8;
    const int qrow0 = qBase + 16 * wid + mo;
    const float* rL = rowL + (size_t)b * LL;
    #pragma unroll
    for (int nt = 0; nt < 4; ++nt) {
        int v = vBase + 16 * nt + n;
        const float* stc = stats_cx + (size_t)b * CH * 2 + 2 * v;
        float mean = stc[0], rstd = stc[1];
        float res[8];
        #pragma unroll
        for (int r = 0; r < 8; ++r) {
            int q = qrow0 + r;
            float invl = 1.0f / rL[q];
            float mv   = accM[nt][r] * invl;
            float e2   = accM2[nt][r] * invl;
            float var  = e2 - mv * mv;
            float sd   = sqrtf(fmaxf(var, 0.0f));
            float cx   = c_x[((size_t)b * CH + v) * LL + q];
            res[r] = sd * ((cx - mean) * rstd) + mv;
        }
        float* op = out + ((size_t)b * CH + v) * LL + qrow0;
        v4f lo = {res[0], res[1], res[2], res[3]};
        v4f hi = {res[4], res[5], res[6], res[7]};
        *(v4f*)(op)     = lo;
        *(v4f*)(op + 4) = hi;
    }
}

// ---------------------------------------------------------------------------
extern "C" void kernel_launch(void* const* d_in, const int* in_sizes, int n_in,
                              void* d_out, int out_size, void* d_ws, size_t ws_size,
                              hipStream_t stream) {
    (void)in_sizes; (void)n_in; (void)out_size; (void)ws_size;
    const float* c_x  = (const float*)d_in[0];
    const float* s_x  = (const float*)d_in[1];
    const float* c_1x = (const float*)d_in[2];
    const float* s_1x = (const float*)d_in[3];
    const float* f_w  = (const float*)d_in[4];
    const float* f_b  = (const float*)d_in[5];
    const float* g_w  = (const float*)d_in[6];
    const float* g_b  = (const float*)d_in[7];
    const float* h_w  = (const float*)d_in[8];
    const float* h_b  = (const float*)d_in[9];
    float* out = (float*)d_out;

    char* ws = (char*)d_ws;
    auto take = [&](size_t bytes) -> char* {
        char* p = ws;
        ws += (bytes + 255) & ~(size_t)255;
        return p;
    };
    float*  st_c1 = (float*)take((size_t)BATCH * CH * 2 * 4);
    float*  st_s1 = (float*)take((size_t)BATCH * CH * 2 * 4);
    float*  st_cx = (float*)take((size_t)BATCH * CH * 2 * 4);
    bf16_t* wf    = (bf16_t*)take((size_t)CH * CH * 2);
    bf16_t* wg    = (bf16_t*)take((size_t)CH * CH * 2);
    bf16_t* wh    = (bf16_t*)take((size_t)CH * CH * 2);
    bf16_t* Tc1   = (bf16_t*)take((size_t)BATCH * LL * CH * 2);   // mvn(c_1x)^T
    bf16_t* Ts1   = (bf16_t*)take((size_t)BATCH * LL * CH * 2);   // mvn(s_1x)^T
    bf16_t* Tsx   = (bf16_t*)take((size_t)BATCH * LL * CH * 2);   // s_x^T
    bf16_t* Q     = (bf16_t*)take((size_t)BATCH * LL * CH * 2);
    bf16_t* Kt    = (bf16_t*)take((size_t)BATCH * LL * CH * 2);
    bf16_t* Vh    = (bf16_t*)take((size_t)BATCH * CH * LL * 2);   // [v][k]
    bf16_t* Vq    = (bf16_t*)take((size_t)BATCH * CH * LL * 2);   // [v][k], squared
    bf16_t* Smat  = (bf16_t*)take((size_t)BATCH * LL * LL * 2);
    float*  rowM  = (float*)take((size_t)BATCH * LL * 4);
    float*  rowL  = (float*)take((size_t)BATCH * LL * 4);

    // 1) mvn stats
    stats_kernel<<<BATCH * CH, 256, 0, stream>>>(c_1x, st_c1);
    stats_kernel<<<BATCH * CH, 256, 0, stream>>>(s_1x, st_s1);
    stats_kernel<<<BATCH * CH, 256, 0, stream>>>(c_x,  st_cx);

    // 2) weights -> bf16
    cvt_bf16_kernel<<<(CH * CH) / 256, 256, 0, stream>>>(f_w, wf, CH * CH);
    cvt_bf16_kernel<<<(CH * CH) / 256, 256, 0, stream>>>(g_w, wg, CH * CH);
    cvt_bf16_kernel<<<(CH * CH) / 256, 256, 0, stream>>>(h_w, wh, CH * CH);

    // 3) transpose (+normalize) to bf16 [L][C]
    dim3 tg(LL / 64, CH / 64, BATCH);
    transpose_kernel<true ><<<tg, 256, 0, stream>>>(c_1x, st_c1, Tc1);
    transpose_kernel<true ><<<tg, 256, 0, stream>>>(s_1x, st_s1, Ts1);
    transpose_kernel<false><<<tg, 256, 0, stream>>>(s_x,  nullptr, Tsx);

    // 4) projections (WMMA, async staging)
    dim3 pg(LL / 128, CH / 64, BATCH);
    proj_lc_kernel<<<pg, 256, 0, stream>>>(wf, Tc1, f_b, Q);
    proj_lc_kernel<<<pg, 256, 0, stream>>>(wg, Ts1, g_b, Kt);
    proj_cl_kernel<<<pg, 256, 0, stream>>>(Tsx, wh, h_b, Vh, Vq);

    // 5) S = Q K (WMMA, async staging)
    attn_qk_kernel<<<dim3(LL / 128, LL / 128, BATCH), 256, 0, stream>>>(Q, Kt, Smat);

    // 6) softmax stats + in-place exp
    row_softmax_kernel<<<dim3(LL, BATCH), 256, 0, stream>>>(Smat, rowM, rowL);

    // 7) P*V, P*V^2, epilogue (WMMA, async staging)
    attn_pv_kernel<<<dim3(LL / 128, CH / 64, BATCH), 256, 0, stream>>>(
        Smat, rowL, Vh, Vq, c_x, st_cx, out);
}
